// Encoder_66065186947370
// MI455X (gfx1250) — matrix operand order repaired and checked by reading, not compiled, not run on hardware
//
#include <hip/hip_runtime.h>
#include <hip/hip_bf16.h>
#include <math.h>

typedef __attribute__((ext_vector_type(16))) __bf16          v16bf;
typedef __attribute__((ext_vector_type(8)))  float           v8f;
typedef __attribute__((ext_vector_type(8)))  unsigned short  v8us;
typedef __attribute__((ext_vector_type(4)))  unsigned int    v4u;

#define LDS_PAD 40   // row stride (elements) for 32-wide bf16 tiles; 80B keeps 16B alignment

__device__ __forceinline__ unsigned short f2bfu(float f) {
    unsigned int u = __float_as_uint(f);
    unsigned int r = u + 0x7FFFu + ((u >> 16) & 1u);   // RNE
    return (unsigned short)(r >> 16);
}

__device__ __forceinline__ v16bf ld_frag(const unsigned short* base, int hi_delta) {
    v8us lo = *(const v8us*)(base);
    v8us hi = *(const v8us*)(base + hi_delta);
    return __builtin_bit_cast(v16bf, __builtin_shufflevector(
        lo, hi, 0,1,2,3,4,5,6,7,8,9,10,11,12,13,14,15));
}

// -------------------------------------------------------------------------
// Weight prep: W (Kc x Nc, fp32) -> Wt (Nc x Kc, bf16, transposed).
// -------------------------------------------------------------------------
__global__ __launch_bounds__(256) void prep_weight(
    const float* __restrict__ W, unsigned short* __restrict__ Wt, int Kc, int Nc)
{
    __shared__ float tile[32][33];
    const int k0 = blockIdx.y * 32, n0 = blockIdx.x * 32;
    const int tx = threadIdx.x, ty = threadIdx.y;      // blockDim (32,8)
    #pragma unroll
    for (int i = ty; i < 32; i += 8)
        tile[i][tx] = W[(long)(k0 + i) * Nc + n0 + tx];
    __syncthreads();
    #pragma unroll
    for (int i = ty; i < 32; i += 8)
        Wt[(long)(n0 + i) * Kc + k0 + tx] = f2bfu(tile[tx][i]);
}

// -------------------------------------------------------------------------
// Elementwise fp32 -> bf16 (level-0 activation prep).
// -------------------------------------------------------------------------
__global__ __launch_bounds__(256) void f32_to_bf16(
    const float* __restrict__ src, unsigned short* __restrict__ dst, long n)
{
    const long i = (long)blockIdx.x * 256 + threadIdx.x;
    if (i < n) dst[i] = f2bfu(src[i]);
}

// -------------------------------------------------------------------------
// Y[M x Nc] = A[M x Kc] @ W[Kc x Nc] + bias; A bf16 row-major, W bf16^T.
// Output fp32 (BF16OUT=false) or bf16 (BF16OUT=true).
// Block: 256 threads = 8 waves. Block tile 128x128, BK=32.
// Wave grid 4x2: wave owns a 32x64 slab = 2x4 WMMA subtiles = 8 WMMAs/K-step.
// Software-pipelined; zero conversion work in the loop.
// Requires M%128==0, Nc%128==0, Kc%32==0 (true for all shapes here).
// -------------------------------------------------------------------------
template <bool BF16OUT>
__global__ __launch_bounds__(256) void gemm_bf16_wmma(
    const unsigned short* __restrict__ A, const unsigned short* __restrict__ Wt,
    const float* __restrict__ bias, float* __restrict__ Yf,
    unsigned short* __restrict__ Ybf, int M, int Nc, int Kc)
{
    __shared__ __align__(16) unsigned short As[128 * LDS_PAD]; // [row][k]
    __shared__ __align__(16) unsigned short Bs[128 * LDS_PAD]; // [col][k]

    const int tid  = threadIdx.x;
    const int wave = tid >> 5, lane = tid & 31;
    const int wr = wave & 3, wc = wave >> 2;       // 4 x 2 wave grid
    const int lm = lane & 15, lh = lane >> 4;
    const int m0 = blockIdx.y * 128;
    const int n0 = blockIdx.x * 128;

    // A staging: 128 rows x 32 K, 16 bf16/thread (contiguous K run)
    const int ar = tid >> 1;             // row 0..127
    const int akq = (tid & 1) * 16;      // k offset 0/16
    // B staging: 128 cols x 32 K, 16 bf16/thread (contiguous in Wt row)
    const int bn = tid & 127;            // col 0..127
    const int bkq = (tid >> 7) * 16;     // k offset 0/16

    const unsigned short* aG = A  + (long)(m0 + ar) * Kc + akq;
    const unsigned short* wG = Wt + (long)(n0 + bn) * Kc + bkq;

    v8f acc[2][4] = {};

    v8us a0, a1, b0, b1;
    // prefetch first K-tile into registers
    a0 = *(const v8us*)(aG);
    a1 = *(const v8us*)(aG + 8);
    b0 = *(const v8us*)(wG);
    b1 = *(const v8us*)(wG + 8);

    for (int kt = 0; kt < Kc; kt += 32) {
        // ---- stage current tile: raw b128 copies, no conversion ----
        {
            v4u* dst = (v4u*)(&As[ar * LDS_PAD + akq]);
            dst[0] = __builtin_bit_cast(v4u, a0);
            dst[1] = __builtin_bit_cast(v4u, a1);
        }
        {
            v4u* dst = (v4u*)(&Bs[bn * LDS_PAD + bkq]);
            dst[0] = __builtin_bit_cast(v4u, b0);
            dst[1] = __builtin_bit_cast(v4u, b1);
        }
        __syncthreads();

        // ---- issue next tile's global loads (overlap with ds_load + WMMA) ----
        const int ktn = kt + 32;
        if (ktn < Kc) {
            a0 = *(const v8us*)(aG + ktn);
            a1 = *(const v8us*)(aG + ktn + 8);
            b0 = *(const v8us*)(wG + ktn);
            b1 = *(const v8us*)(wG + ktn + 8);
        }

        // ---- fragments (ISA 7.12.2 16-bit layouts) ----
        // A 16x32: lane lm = M row; lanes0-15 K0-7/16-23, lanes16-31 K8-15/24-31
        v16bf af[2];
        #pragma unroll
        for (int i = 0; i < 2; ++i)
            af[i] = ld_frag(&As[(wr * 32 + i * 16 + lm) * LDS_PAD + lh * 8], 16);
        // B 32x16: lane lm = N col; lanes0-15 K0-7/16-23, lanes16-31 K8-15/24-31
        v16bf bf[4];
        #pragma unroll
        for (int j = 0; j < 4; ++j)
            bf[j] = ld_frag(&Bs[(wc * 64 + j * 16 + lm) * LDS_PAD + lh * 8], 16);

        #pragma unroll
        for (int i = 0; i < 2; ++i)
            #pragma unroll
            for (int j = 0; j < 4; ++j)
                acc[i][j] = __builtin_amdgcn_wmma_f32_16x16x32_bf16(
                    false, af[i], false, bf[j], (short)0, acc[i][j], false, false);
        __syncthreads();
    }

    // ---- store D: VGPR r -> M = r + 8*lh ; N = lm ----
    #pragma unroll
    for (int j = 0; j < 4; ++j) {
        const int gn = n0 + wc * 64 + j * 16 + lm;
        const float bv = bias[gn];                     // never null (zero buffer used)
        #pragma unroll
        for (int i = 0; i < 2; ++i) {
            const long rbase = (long)(m0 + wr * 32 + i * 16 + lh * 8) * Nc + gn;
            #pragma unroll
            for (int r = 0; r < 8; ++r) {
                const float v = acc[i][j][r] + bv;
                if (BF16OUT) Ybf[rbase + r * Nc] = f2bfu(v);
                else         Yf [rbase + r * Nc] = v;
            }
        }
    }
}

// -------------------------------------------------------------------------
// Neighbor attention: one wave per (b,n) row. K = 8 wrap-around neighbors.
// Writes bf16 output (feeds the wo GEMM's A side directly).
// -------------------------------------------------------------------------
__global__ __launch_bounds__(256) void attn_kernel(
    const float* __restrict__ Q, const float* __restrict__ K,
    const float* __restrict__ V, const float* __restrict__ rb,
    unsigned short* __restrict__ T, int Nn, int d, float rsq)
{
    const int wave = threadIdx.x >> 5, lane = threadIdx.x & 31;
    const long row = (long)blockIdx.x * 8 + wave;       // in [0, 2*Nn)
    const int  n   = (int)(row & (long)(Nn - 1));       // Nn is a power of two
    const long bb  = row - n;                           // b * Nn

    const int nc = d >> 5;                              // channels per lane (4/8/16)
    const float* q = Q + row * d;
    float qr[16];
    #pragma unroll
    for (int j = 0; j < 16; ++j) qr[j] = (j < nc) ? q[lane + j * 32] : 0.f;

    float s[8];
    #pragma unroll
    for (int k = 0; k < 8; ++k) {
        const float* kp = K + (bb + ((n + k) & (Nn - 1))) * d;
        float acc = 0.f;
        for (int j = 0; j < nc; ++j) acc += qr[j] * kp[lane + j * 32];
        #pragma unroll
        for (int off = 16; off > 0; off >>= 1) acc += __shfl_xor(acc, off, 32);
        s[k] = (acc + rb[k]) * rsq;
    }
    float mx = s[0];
    #pragma unroll
    for (int k = 1; k < 8; ++k) mx = fmaxf(mx, s[k]);
    float sum = 0.f;
    #pragma unroll
    for (int k = 0; k < 8; ++k) { s[k] = __expf(s[k] - mx); sum += s[k]; }
    const float inv = 1.f / sum;
    #pragma unroll
    for (int k = 0; k < 8; ++k) s[k] *= inv;

    for (int j = 0; j < nc; ++j) {
        const int c = lane + j * 32;
        float o = 0.f;
        #pragma unroll
        for (int k = 0; k < 8; ++k)
            o += s[k] * V[(bb + ((n + k) & (Nn - 1))) * d + c];
        T[row * d + c] = f2bfu(o);
    }
}

// -------------------------------------------------------------------------
// 2x2 patch-merge gather: S (B,H,H,d) fp32 -> MG (B, H/2*H/2, 4d) bf16.
// -------------------------------------------------------------------------
__global__ __launch_bounds__(256) void merge_gather(
    const float* __restrict__ S, unsigned short* __restrict__ MG,
    int H2, int d, long total)
{
    long i = (long)blockIdx.x * 256 + threadIdx.x;
    if (i >= total) return;
    const int c = (int)(i % d);
    long r = i / d;
    const int quad = (int)(r & 3); r >>= 2;
    const int w2 = (int)(r % H2); r /= H2;
    const int h2 = (int)(r % H2);
    const int b  = (int)(r / H2);
    const int H  = H2 * 2;
    const int dh = quad & 1, dw = quad >> 1;  // x0:(0,0) x1:(1,0) x2:(0,1) x3:(1,1)
    const long src = (((long)b * H + (2 * h2 + dh)) * H + (2 * w2 + dw)) * (long)d + c;
    MG[i] = f2bfu(S[src]);
}

extern "C" void kernel_launch(void* const* d_in, const int* in_sizes, int n_in,
                              void* d_out, int out_size, void* d_ws, size_t ws_size,
                              hipStream_t stream)
{
    (void)in_sizes; (void)n_in; (void)out_size; (void)ws_size;
    float* out = (float*)d_out;
    float* ws  = (float*)d_ws;

    const long BUF = 4194304;                 // max per-level activation elements
    float* Qb = ws;                           // fp32
    float* Kb = ws + 1 * BUF;                 // fp32
    float* Vb = ws + 2 * BUF;                 // fp32
    unsigned short* Tbf = (unsigned short*)(ws + 3 * BUF);        // bf16: attn out, then MG
    unsigned short* Xbf = (unsigned short*)(ws + 3 * BUF + BUF/2);// bf16 activations
    float* zb = ws + 3 * BUF + BUF;           // 1024-float zero bias
    unsigned short* wt = (unsigned short*)(zb + 1024);            // bf16 transposed weights

    hipMemsetAsync(zb, 0, 1024 * sizeof(float), stream);   // capture-legal, deterministic

    // d_out layout: [final 2*256*1024][skip0 2*16384*128][skip1 2*4096*256][skip2 2*1024*512]
    const long skip_off[3] = {524288L, 524288L + 4194304L, 524288L + 4194304L + 2097152L};

    const dim3 blk(256);
    const dim3 tblk(32, 8);

    // ---- one-time weight conversion/transpose to bf16 ----
    unsigned short* wqT[3]; unsigned short* wkT[3]; unsigned short* wvT[3];
    unsigned short* woT[3]; unsigned short* wmT[3];
    {
        long off = 0;
        for (int l = 0; l < 3; ++l) {
            const int d = 128 << l;
            const int base = 1 + 10 * l;
            unsigned short* p[5];
            const long sz[5] = {(long)d*d, (long)d*d, (long)d*d, (long)d*d, (long)8*d*d};
            for (int t = 0; t < 5; ++t) { p[t] = wt + off; off += sz[t]; }
            wqT[l]=p[0]; wkT[l]=p[1]; wvT[l]=p[2]; woT[l]=p[3]; wmT[l]=p[4];
            prep_weight<<<dim3(d/32, d/32), tblk, 0, stream>>>(
                (const float*)d_in[base+0], wqT[l], d, d);
            prep_weight<<<dim3(d/32, d/32), tblk, 0, stream>>>(
                (const float*)d_in[base+1], wkT[l], d, d);
            prep_weight<<<dim3(d/32, d/32), tblk, 0, stream>>>(
                (const float*)d_in[base+2], wvT[l], d, d);
            prep_weight<<<dim3(d/32, d/32), tblk, 0, stream>>>(
                (const float*)d_in[base+3], woT[l], d, d);
            prep_weight<<<dim3((2*d)/32, (4*d)/32), tblk, 0, stream>>>(
                (const float*)d_in[base+9], wmT[l], 4*d, 2*d);
        }
    }

    // level-0 activations -> bf16 (consumed by 3 GEMMs)
    f32_to_bf16<<<dim3((unsigned)(BUF / 256)), blk, 0, stream>>>(
        (const float*)d_in[0], Xbf, BUF);

    for (int l = 0; l < 3; ++l) {
        const int d  = 128 << l;
        const int Nn = 16384 >> (2 * l);
        const int M  = 2 * Nn;
        const int H2 = 64 >> l;
        const int base = 1 + 10 * l;
        const float* bq  = (const float*)d_in[base + 4];
        const float* bk  = (const float*)d_in[base + 5];
        const float* bvp = (const float*)d_in[base + 6];
        const float* bo  = (const float*)d_in[base + 7];
        const float* rb  = (const float*)d_in[base + 8]; // rel_bias, row 0 = first 8 floats
        float* skip = out + skip_off[l];

        const dim3 gqkv(d / 128, M / 128);
        gemm_bf16_wmma<false><<<gqkv, blk, 0, stream>>>(Xbf, wqT[l], bq,  Qb, nullptr, M, d, d);
        gemm_bf16_wmma<false><<<gqkv, blk, 0, stream>>>(Xbf, wkT[l], bk,  Kb, nullptr, M, d, d);
        gemm_bf16_wmma<false><<<gqkv, blk, 0, stream>>>(Xbf, wvT[l], bvp, Vb, nullptr, M, d, d);

        attn_kernel<<<dim3(M / 8), blk, 0, stream>>>(Qb, Kb, Vb, rb, Tbf, Nn, d,
                                                     1.0f / sqrtf((float)d));

        gemm_bf16_wmma<false><<<gqkv, blk, 0, stream>>>(Tbf, woT[l], bo, skip, nullptr, M, d, d);

        const long total = (long)M * d;       // == (M/4) * 4d
        merge_gather<<<dim3((unsigned)((total + 255) / 256)), blk, 0, stream>>>(
            skip, Tbf, H2, d, total);         // Tbf reused as bf16 MG

        const dim3 gmg((2 * d) / 128, (M / 4) / 128);
        if (l == 2) {
            gemm_bf16_wmma<false><<<gmg, blk, 0, stream>>>(
                Tbf, wmT[l], zb, out, nullptr, M / 4, 2 * d, 4 * d);
        } else {
            gemm_bf16_wmma<true><<<gmg, blk, 0, stream>>>(
                Tbf, wmT[l], zb, nullptr, Xbf, M / 4, 2 * d, 4 * d);
        }
    }
}
